// SS_SA_14096082665922
// MI455X (gfx1250) — matrix operand order repaired and checked by reading, compiled, not run on hardware
//
#include <hip/hip_runtime.h>
#include <hip/hip_bf16.h>

// MI455X (gfx1250) implementation.
// Pipeline: qkv GEMM (bf16 WMMA, async global->LDS A-tile staging) ->
// depthwise 3x3 -> row l2-norms -> per-(b,head) attention (TDM-staged q/k/v
// tiles, WMMA qk^T, bitonic top-k, 4x sparse softmax, WMMA spar*v) ->
// projection GEMM.
// Workspace: qkv (100.7MB) + qkv_dw (100.7MB) + attn_out (33.6MB) + norms.

typedef __attribute__((ext_vector_type(4)))  float    v4f;
typedef __attribute__((ext_vector_type(8)))  float    v8f;
typedef __attribute__((ext_vector_type(8)))  __bf16   v8bf;
typedef __attribute__((ext_vector_type(16))) __bf16   v16bf;
typedef __attribute__((ext_vector_type(4)))  unsigned u32x4;
typedef __attribute__((ext_vector_type(8)))  unsigned u32x8;

union Frag16 { v16bf v; v8bf h[2]; };

// Low 32 bits of a generic pointer to __shared__ = LDS byte address (ISA 10.2).
__device__ __forceinline__ unsigned lds_off(const void* p) {
  return (unsigned)(size_t)p;
}

// GLOBAL_LOAD_ASYNC_TO_LDS_B128: per-lane 16B global->LDS copy, ASYNCcnt.
__device__ __forceinline__ void async_copy_b128(const void* gsrc, void* ldst) {
  unsigned l = lds_off(ldst);
  asm volatile("global_load_async_to_lds_b128 %0, %1, off"
               :: "v"(l), "v"(gsrc) : "memory");
}
__device__ __forceinline__ void wait_async0() {
  asm volatile("s_wait_asynccnt 0x0" ::: "memory");
}

// TENSOR_LOAD_TO_LDS with a 2D descriptor (D# groups 0/1, ISA 8.3/8.4).
// Loads tile_d1 rows of tile_d0 elements (f32) from a tensor with row stride
// stride0 (elements) into LDS row-major, tracked by TENSORcnt.
__device__ __forceinline__ void tdm_load_2d(unsigned lds_addr, const void* gaddr,
                                            unsigned tensor_d0, unsigned tensor_d1,
                                            unsigned tile_d0, unsigned tile_d1,
                                            unsigned stride0) {
  unsigned long long ga = (unsigned long long)(size_t)gaddr;
  u32x4 g0;
  g0[0] = 1u;                                   // count=1, user mode
  g0[1] = lds_addr;                             // lds_addr [63:32]
  g0[2] = (unsigned)ga;                         // global_addr [95:64]
  g0[3] = (unsigned)(ga >> 32) | (2u << 30);    // global_addr hi | type=2
  u32x8 g1;
  g1[0] = 0x20000u;                             // data_size=2 (4B), mask=0
  g1[1] = (tensor_d0 & 0xFFFFu) << 16;          // tensor_dim0 [79:48]
  g1[2] = (tensor_d0 >> 16) | ((tensor_d1 & 0xFFFFu) << 16);  // dim0 hi | dim1 lo
  g1[3] = (tensor_d1 >> 16) | (tile_d0 << 16);  // dim1 hi | tile_dim0 [127:112]
  g1[4] = tile_d1;                              // tile_dim1, tile_dim2=0
  g1[5] = stride0;                              // tensor_dim0_stride lo32
  g1[6] = 0u;                                   // stride0 hi | stride1 lo
  g1[7] = 0u;
  asm volatile("tensor_load_to_lds %0, %1" :: "s"(g0), "s"(g1) : "memory");
}

#define BM 128
#define BN 64
#define BK 32

// ---------------------------------------------------------------------------
// GEMM: Y[b][M][N] = W[M][K] * X[b][K][N], K=256, N=16384, f32 in/out,
// bf16 WMMA 16x16x32; A tile staged f32 via async global->LDS copies.
// ---------------------------------------------------------------------------
__global__ __launch_bounds__(256) void gemm_bf16_wmma(
    const float* __restrict__ Wt,   // [M][256]
    const float* __restrict__ X,    // [batch][256][16384]
    float* __restrict__ Y,          // [batch][M][16384]
    int M)
{
  const int K = 256, N = 16384;
  const int b  = blockIdx.z;
  const int m0 = blockIdx.y * BM;
  const int n0 = blockIdx.x * BN;
  const float* Xb = X + (size_t)b * K * N;
  float*       Yb = Y + (size_t)b * M * N;

  __shared__ float  Asf[2][BM][BK];   // 16KB x2: raw f32 weights (async-staged)
  __shared__ __bf16 Bs[2][BN][BK];    // 4KB x2: bf16 activations, [n][k]

  const int tid = threadIdx.x;
  const int wid = tid >> 5;
  const int lane = tid & 31;
  const int laneHalf = lane >> 4;
  const int laneRow  = lane & 15;
  const int wm = wid & 3;
  const int wn = wid >> 2;

  v8f acc[2][2] = {};

  auto stage = [&](int buf, int kt) {
    const int k0 = kt * BK;
    // A tile: 128x32 f32, 4 async 16B copies per thread (no VGPR round-trip)
    #pragma unroll
    for (int r = 0; r < 4; ++r) {
      int idx = (r * 256 + tid) * 4;
      int row = idx >> 5;
      int col = idx & 31;
      async_copy_b128(Wt + (size_t)(m0 + row) * K + k0 + col,
                      &Asf[buf][row][col]);
    }
    // B tile: 32x64 f32 -> bf16 transposed into [n][k]
    #pragma unroll
    for (int r = 0; r < 2; ++r) {
      int idx = (r * 256 + tid) * 4;
      int kk = idx >> 6;
      int nn = idx & 63;
      const float* src = Xb + (size_t)(k0 + kk) * N + n0 + nn;
      __builtin_prefetch(src + BK * N, 0, 0);   // global_prefetch_b8
      v4f xv = *(const v4f*)src;
      Bs[buf][nn + 0][kk] = (__bf16)xv[0];
      Bs[buf][nn + 1][kk] = (__bf16)xv[1];
      Bs[buf][nn + 2][kk] = (__bf16)xv[2];
      Bs[buf][nn + 3][kk] = (__bf16)xv[3];
    }
  };

  stage(0, 0);
  const int KT = K / BK;   // 8
  for (int kt = 0; kt < KT; ++kt) {
    wait_async0();          // this wave's async copies landed in LDS
    __syncthreads();        // all waves' staging visible
    if (kt + 1 < KT) stage((kt + 1) & 1, kt + 1);
    const int buf = kt & 1;

    Frag16 af[2], bfq[2];
    #pragma unroll
    for (int i = 0; i < 2; ++i) {               // A frag built from f32 LDS
      int row = wm * 32 + i * 16 + laneRow;
      const float* ap = &Asf[buf][row][0];
      v4f a0 = *(const v4f*)(ap + laneHalf * 8);
      v4f a1 = *(const v4f*)(ap + laneHalf * 8 + 4);
      v4f a2 = *(const v4f*)(ap + 16 + laneHalf * 8);
      v4f a3 = *(const v4f*)(ap + 16 + laneHalf * 8 + 4);
      #pragma unroll
      for (int e = 0; e < 4; ++e) {
        af[i].v[e]      = (__bf16)a0[e];
        af[i].v[4 + e]  = (__bf16)a1[e];
        af[i].v[8 + e]  = (__bf16)a2[e];
        af[i].v[12 + e] = (__bf16)a3[e];
      }
    }
    #pragma unroll
    for (int j = 0; j < 2; ++j) {
      int col = wn * 32 + j * 16 + laneRow;
      bfq[j].h[0] = *(const v8bf*)&Bs[buf][col][laneHalf * 8];
      bfq[j].h[1] = *(const v8bf*)&Bs[buf][col][16 + laneHalf * 8];
    }
    #pragma unroll
    for (int i = 0; i < 2; ++i)
      #pragma unroll
      for (int j = 0; j < 2; ++j)
        acc[i][j] = __builtin_amdgcn_wmma_f32_16x16x32_bf16(
            false, af[i].v, false, bfq[j].v, (short)0, acc[i][j], false, false);
  }

  #pragma unroll
  for (int i = 0; i < 2; ++i) {
    int rowBase = m0 + wm * 32 + i * 16 + laneHalf * 8;
    #pragma unroll
    for (int j = 0; j < 2; ++j) {
      int col = n0 + wn * 32 + j * 16 + laneRow;
      #pragma unroll
      for (int r = 0; r < 8; ++r)
        Yb[(size_t)(rowBase + r) * N + col] = acc[i][j][r];
    }
  }
}

// ---------------------------------------------------------------------------
// Depthwise 3x3 'SAME' conv (768 groups), memory-bound (L2-resident).
// ---------------------------------------------------------------------------
__global__ __launch_bounds__(256) void dwconv3x3(
    const float* __restrict__ In, const float* __restrict__ Wd,
    float* __restrict__ Out)
{
  const int N = 16384, Wimg = 128;
  const int b = blockIdx.z;
  const int o = blockIdx.y;
  const int p = blockIdx.x * 256 + threadIdx.x;
  const int y = p >> 7, x = p & 127;
  const float* ip = In + ((size_t)b * 768 + o) * N;
  float wv[9];
  #pragma unroll
  for (int i = 0; i < 9; ++i) wv[i] = Wd[o * 9 + i];
  float s = 0.f;
  #pragma unroll
  for (int ky = -1; ky <= 1; ++ky) {
    int yy = y + ky;
    if (yy < 0 || yy > 127) continue;
    #pragma unroll
    for (int kx = -1; kx <= 1; ++kx) {
      int xx = x + kx;
      if (xx < 0 || xx > 127) continue;
      s += wv[(ky + 1) * 3 + (kx + 1)] * ip[yy * Wimg + xx];
    }
  }
  Out[((size_t)b * 768 + o) * N + p] = s;
}

// ---------------------------------------------------------------------------
// Per-row inverse L2 norms for q (rows 0..511) and k (rows 512..1023).
// ---------------------------------------------------------------------------
__global__ __launch_bounds__(256) void rownorm(
    const float* __restrict__ QKV, float* __restrict__ rnq,
    float* __restrict__ rnk)
{
  const int idx   = blockIdx.x;
  const int which = idx >> 9;
  const int rem   = idx & 511;
  const int b  = rem >> 8;
  const int ch = rem & 255;
  const float* row = QKV + ((size_t)b * 768 + which * 256 + ch) * 16384;
  float s = 0.f;
  for (int i = threadIdx.x; i < 16384; i += 256) { float v = row[i]; s += v * v; }
  __shared__ float red[256];
  red[threadIdx.x] = s;
  __syncthreads();
  for (int st = 128; st > 0; st >>= 1) {
    if (threadIdx.x < st) red[threadIdx.x] += red[threadIdx.x + st];
    __syncthreads();
  }
  if (threadIdx.x == 0) {
    float inv = 1.f / fmaxf(sqrtf(red[0]), 1e-12f);
    (which ? rnk : rnq)[(size_t)b * 256 + ch] = inv;
  }
}

// ---------------------------------------------------------------------------
// Attention per (b, head). q/k/v tiles staged by the Tensor Data Mover with
// TENSORcnt double-buffering; WMMA qk^T; exact top-k via bitonic sort; 4x
// sparse softmax blend; WMMA spar*v.
// ---------------------------------------------------------------------------
__global__ __launch_bounds__(256) void attn_kernel(
    const float* __restrict__ QKV,  // dw output [b][768][HW]
    const float* __restrict__ rnq, const float* __restrict__ rnk,
    const float* __restrict__ temp,
    const float* __restrict__ a1, const float* __restrict__ a2,
    const float* __restrict__ a3, const float* __restrict__ a4,
    float* __restrict__ Out)        // [b][256][HW]
{
  const int HW = 16384;
  const int b = blockIdx.x >> 3;
  const int h = blockIdx.x & 7;
  const float* q = QKV + ((size_t)b * 768 + h * 32) * HW;
  const float* k = QKV + ((size_t)b * 768 + 256 + h * 32) * HW;
  const float* v = QKV + ((size_t)b * 768 + 512 + h * 32) * HW;
  const float* rq = rnq + (size_t)b * 256 + h * 32;
  const float* rk = rnk + (size_t)b * 256 + h * 32;

  const int tid = threadIdx.x;
  const int wid = tid >> 5;
  const int lane = tid & 31;
  const int laneHalf = lane >> 4;
  const int laneRow  = lane & 15;

  __shared__ float  qs[2][8][32][32];  // 64KB: per-wave TDM-staged q tiles
  __shared__ float  kt[2][8][32][32];  // 64KB: per-wave TDM-staged k tiles
  __shared__ float  vs[2][8][32][16];  // 32KB: per-wave TDM-staged v tiles
  __shared__ float  part[8][32][32];   // 32KB: per-wave partials / reductions
  __shared__ float  attnS[1024];
  __shared__ float  sortS[1024];
  __shared__ __bf16 sparS[32][32];

  const unsigned qAddr[2] = { lds_off(&qs[0][wid][0][0]), lds_off(&qs[1][wid][0][0]) };
  const unsigned kAddr[2] = { lds_off(&kt[0][wid][0][0]), lds_off(&kt[1][wid][0][0]) };
  const unsigned vAddr[2] = { lds_off(&vs[0][wid][0][0]), lds_off(&vs[1][wid][0][0]) };

  // ---- Phase 1: attn partials, each wave reduces 2048 of the 16384 K dim ----
  v8f acc[2][2] = {};
  const int kbase = wid * 2048;
  // prologue: TDM-load first q/k tiles (32 rows x 32 cols, row stride HW)
  tdm_load_2d(qAddr[0], q + kbase, HW, 32, 32, 32, HW);
  tdm_load_2d(kAddr[0], k + kbase, HW, 32, 32, 32, HW);
  int buf = 0;
  for (int ks = 0; ks < 2048; ks += 32) {
    if (ks + 32 < 2048) {
      const int kn = kbase + ks + 32;
      tdm_load_2d(qAddr[buf ^ 1], q + kn, HW, 32, 32, 32, HW);
      tdm_load_2d(kAddr[buf ^ 1], k + kn, HW, 32, 32, 32, HW);
      __builtin_amdgcn_s_wait_tensorcnt((short)2);  // current pair landed
    } else {
      __builtin_amdgcn_s_wait_tensorcnt((short)0);
    }
    const float* qb = &qs[buf][wid][0][0];
    const float* kb = &kt[buf][wid][0][0];

    Frag16 af[2], bfq[2];
    #pragma unroll
    for (int i = 0; i < 2; ++i) {
      int row = i * 16 + laneRow;
      float sc = rq[row];
      const float* p0 = qb + row * 32 + laneHalf * 16;
      const float* p1 = p0 + 8;
      #pragma unroll
      for (int e = 0; e < 8; ++e) {
        af[i].v[e]     = (__bf16)(p0[e] * sc);
        af[i].v[8 + e] = (__bf16)(p1[e] * sc);
      }
    }
    #pragma unroll
    for (int j = 0; j < 2; ++j) {
      int col = j * 16 + laneRow;
      float sc = rk[col];
      const float* p0 = kb + col * 32 + laneHalf * 16;
      #pragma unroll
      for (int e = 0; e < 16; ++e) bfq[j].v[e] = (__bf16)(p0[e] * sc);
    }
    #pragma unroll
    for (int i = 0; i < 2; ++i)
      #pragma unroll
      for (int j = 0; j < 2; ++j)
        acc[i][j] = __builtin_amdgcn_wmma_f32_16x16x32_bf16(
            false, af[i].v, false, bfq[j].v, (short)0, acc[i][j], false, false);
    buf ^= 1;
  }
  #pragma unroll
  for (int i = 0; i < 2; ++i)
    #pragma unroll
    for (int j = 0; j < 2; ++j)
      #pragma unroll
      for (int r = 0; r < 8; ++r)
        part[wid][i * 16 + laneHalf * 8 + r][j * 16 + laneRow] = acc[i][j][r];
  __syncthreads();

  // cross-wave reduce + temperature
  const float tscale = temp[h];
  for (int e = tid; e < 1024; e += 256) {
    int r = e >> 5, cc = e & 31;
    float s = 0.f;
    #pragma unroll
    for (int wv = 0; wv < 8; ++wv) s += part[wv][r][cc];
    s *= tscale;
    attnS[e] = s;
    sortS[e] = s;
  }
  __syncthreads();

  // ---- Bitonic sort (ascending): exact top-k thresholds --------------------
  for (int kk = 2; kk <= 1024; kk <<= 1) {
    for (int j = kk >> 1; j > 0; j >>= 1) {
      for (int t = tid; t < 1024; t += 256) {
        int ixj = t ^ j;
        if (ixj > t) {
          float x0 = sortS[t], x1 = sortS[ixj];
          bool up = ((t & kk) == 0);
          if ((x0 > x1) == up) { sortS[t] = x1; sortS[ixj] = x0; }
        }
      }
      __syncthreads();
    }
  }

  const float maxv = sortS[1023];
  const float th0 = sortS[1024 - 512];   // top 1/2
  const float th1 = sortS[1024 - 682];   // top 2/3
  const float th2 = sortS[1024 - 768];   // top 3/4
  const float th3 = sortS[1024 - 819];   // top 4/5

  float ps[4] = {0.f, 0.f, 0.f, 0.f};
  for (int e = tid; e < 1024; e += 256) {
    float x0 = attnS[e];
    float ex = __expf(x0 - maxv);
    if (x0 >= th0) ps[0] += ex;
    if (x0 >= th1) ps[1] += ex;
    if (x0 >= th2) ps[2] += ex;
    if (x0 >= th3) ps[3] += ex;
  }
  float* red = &part[0][0][0];
  #pragma unroll
  for (int m = 0; m < 4; ++m) red[tid * 4 + m] = ps[m];
  __syncthreads();
  for (int st = 128; st > 0; st >>= 1) {
    if (tid < st) {
      #pragma unroll
      for (int m = 0; m < 4; ++m) red[tid * 4 + m] += red[(tid + st) * 4 + m];
    }
    __syncthreads();
  }
  const float w0 = a1[0] / red[0];
  const float w1 = a2[0] / red[1];
  const float w2 = a3[0] / red[2];
  const float w3 = a4[0] / red[3];

  for (int e = tid; e < 1024; e += 256) {
    float x0 = attnS[e];
    float ex = __expf(x0 - maxv);
    float sp = 0.f;
    if (x0 >= th0) sp += w0 * ex;
    if (x0 >= th1) sp += w1 * ex;
    if (x0 >= th2) sp += w2 * ex;
    if (x0 >= th3) sp += w3 * ex;
    sparS[e >> 5][e & 31] = (__bf16)sp;
  }
  __syncthreads();

  // ---- Phase 2: out = spar(32x32) * v(32xHW), TDM-staged v tiles -----------
  Frag16 af2[2];
  #pragma unroll
  for (int i = 0; i < 2; ++i) {
    int row = i * 16 + laneRow;
    af2[i].h[0] = *(const v8bf*)&sparS[row][laneHalf * 16];
    af2[i].h[1] = *(const v8bf*)&sparS[row][laneHalf * 16 + 8];
  }
  float* outp = Out + ((size_t)b * 256 + h * 32) * HW;
  const int nFirst = wid * 16;
  tdm_load_2d(vAddr[0], v + nFirst, HW, 32, 16, 32, HW);  // 32 rows x 16 cols
  int buf2 = 0;
  for (int n0 = nFirst; n0 < HW; n0 += 128) {
    if (n0 + 128 < HW) {
      tdm_load_2d(vAddr[buf2 ^ 1], v + n0 + 128, HW, 32, 16, 32, HW);
      __builtin_amdgcn_s_wait_tensorcnt((short)1);
    } else {
      __builtin_amdgcn_s_wait_tensorcnt((short)0);
    }
    const float* vb = &vs[buf2][wid][0][0];
    Frag16 bf2;
    #pragma unroll
    for (int e = 0; e < 16; ++e)
      bf2.v[e] = (__bf16)vb[(laneHalf * 16 + e) * 16 + laneRow];
    v8f o0 = {}, o1 = {};
    o0 = __builtin_amdgcn_wmma_f32_16x16x32_bf16(
        false, af2[0].v, false, bf2.v, (short)0, o0, false, false);
    o1 = __builtin_amdgcn_wmma_f32_16x16x32_bf16(
        false, af2[1].v, false, bf2.v, (short)0, o1, false, false);
    const int colN = n0 + laneRow;
    #pragma unroll
    for (int r = 0; r < 8; ++r) {
      outp[(size_t)(laneHalf * 8 + r) * HW + colN]      = o0[r];
      outp[(size_t)(16 + laneHalf * 8 + r) * HW + colN] = o1[r];
    }
    buf2 ^= 1;
  }
}

// ---------------------------------------------------------------------------
extern "C" void kernel_launch(void* const* d_in, const int* in_sizes, int n_in,
                              void* d_out, int out_size, void* d_ws, size_t ws_size,
                              hipStream_t stream) {
  const float* x      = (const float*)d_in[0];
  const float* w_qkv  = (const float*)d_in[1];
  const float* w_dw   = (const float*)d_in[2];
  const float* w_proj = (const float*)d_in[3];
  const float* temperature = (const float*)d_in[4];
  const float* a1 = (const float*)d_in[5];
  const float* a2 = (const float*)d_in[6];
  const float* a3 = (const float*)d_in[7];
  const float* a4 = (const float*)d_in[8];
  float* out = (float*)d_out;

  float* ws = (float*)d_ws;
  const size_t qkvN = (size_t)2 * 768 * 16384;
  float* qkv      = ws;
  float* qkvdw    = qkv + qkvN;
  float* attn_out = qkvdw + qkvN;
  float* rnq      = attn_out + (size_t)2 * 256 * 16384;
  float* rnk      = rnq + 512;

  gemm_bf16_wmma<<<dim3(16384 / BN, 768 / BM, 2), 256, 0, stream>>>(
      w_qkv, x, qkv, 768);
  dwconv3x3<<<dim3(16384 / 256, 768, 2), 256, 0, stream>>>(qkv, w_dw, qkvdw);
  rownorm<<<dim3(1024), 256, 0, stream>>>(qkvdw, rnq, rnk);
  attn_kernel<<<dim3(16), 256, 0, stream>>>(
      qkvdw, rnq, rnk, temperature, a1, a2, a3, a4, attn_out);
  gemm_bf16_wmma<<<dim3(16384 / BN, 256 / BM, 2), 256, 0, stream>>>(
      w_proj, attn_out, out, 256);
}
